// SNNModel_63745904608017
// MI455X (gfx1250) — matrix operand order repaired and checked by reading, not compiled, not run on hardware
//
#include <hip/hip_runtime.h>

// ---------------------------------------------------------------------------
// SNN LIF pipeline for MI455X (gfx1250):
//   [1] transpose-convert stimulus -> A_bf16[t][i] = in[i][t+1]   (32 MB)
//   [2] convert weights            -> W_bf16[o][i]                (32 MB)
//   [3] bf16 WMMA GEMM (f32 accum) -> rf[t][o]                    (64 MB)
//   [4] sequential LIF scan per neuron -> spikes out[o][t]
// Workspace requirement: 128 MB. Both bf16 operands (64 MB) fit in the
// 192 MB L2, so the GEMM loads fragments straight from global (L2-resident)
// instead of staging through LDS.
// ---------------------------------------------------------------------------

typedef __attribute__((ext_vector_type(4)))  __bf16 v4bf;
typedef __attribute__((ext_vector_type(8)))  __bf16 v8bf;
typedef __attribute__((ext_vector_type(16))) __bf16 v16bf;
typedef __attribute__((ext_vector_type(4)))  float  v4f;
typedef __attribute__((ext_vector_type(8)))  float  v8f;

#define N_IN    4096
#define N_OUT   4096
#define T_TOT   4096
#define T_STEPS 4095   // reference records T-1 steps

// ---------------------------------------------------------------- kernel 1 --
// A[t][i] = in[i][t+1]  (bf16, zero for t+1 >= T_TOT i.e. the pad row 4095)
__global__ void snn_build_A(const float* __restrict__ in, __bf16* __restrict__ A) {
  __shared__ float tile[32][33];           // +1 pad: conflict-free transpose
  const int t0 = blockIdx.x * 32;
  const int i0 = blockIdx.y * 32;
  const int tx = threadIdx.x, ty = threadIdx.y;
#pragma unroll
  for (int r = 0; r < 4; ++r) {
    const int i = i0 + ty + r * 8;
    const int t = t0 + 1 + tx;             // shifted column (t+1)
    tile[ty + r * 8][tx] = (t < T_TOT) ? in[(size_t)i * T_TOT + t] : 0.0f;
  }
  __syncthreads();
#pragma unroll
  for (int r = 0; r < 4; ++r) {
    const int t = t0 + ty + r * 8;
    const int i = i0 + tx;
    A[(size_t)t * N_IN + i] = (__bf16)tile[tx][ty + r * 8];
  }
}

// ---------------------------------------------------------------- kernel 2 --
__global__ void snn_convert_w(const float* __restrict__ w, __bf16* __restrict__ wb) {
  const size_t i = ((size_t)blockIdx.x * blockDim.x + threadIdx.x) * 4;
  v4f x = *(const v4f*)(w + i);
  v4bf y;
  y[0] = (__bf16)x[0]; y[1] = (__bf16)x[1];
  y[2] = (__bf16)x[2]; y[3] = (__bf16)x[3];
  *(v4bf*)(wb + i) = y;
}

// ---------------------------------------------------------------- kernel 3 --
// rf[t][o] = sum_i A[t][i] * W[o][i], via v_wmma_f32_16x16x32_bf16.
// Block = 8 waves (256 thr), wave tile 64x64, block tile 256(M) x 128(N).
__global__ __launch_bounds__(256) void snn_gemm_bf16(
    const __bf16* __restrict__ A, const __bf16* __restrict__ B,
    float* __restrict__ rf) {
  const int lane  = threadIdx.x & 31;
  const int wave  = threadIdx.x >> 5;
  const int wm    = wave >> 1;                 // 0..3 (M dir)
  const int wn    = wave & 1;                  // 0..1 (N dir)
  const int mBase = blockIdx.y * 256 + wm * 64;
  const int nBase = blockIdx.x * 128 + wn * 64;
  const int l15   = lane & 15;
  const int lhi   = lane >> 4;

  const v8f zero = {0.f, 0.f, 0.f, 0.f, 0.f, 0.f, 0.f, 0.f};
  v8f c[4][4];
#pragma unroll
  for (int mi = 0; mi < 4; ++mi)
#pragma unroll
    for (int ni = 0; ni < 4; ++ni) c[mi][ni] = zero;

  // Per-lane row bases. A fragment (16x32 bf16): lane = M row, lane-half picks
  // K chunks {0..7,16..23} vs {8..15,24..31}. B fragment (32x16 bf16): lane =
  // N column, lane-half picks K chunks {0..7,16..23} vs {8..15,24..31}.
  const __bf16* aRow = A + (size_t)(mBase + l15) * N_IN;
  const __bf16* bRow = B + (size_t)(nBase + l15) * N_IN;

  for (int kb = 0; kb < N_IN; kb += 32) {
    v16bf afrag[4], bfrag[4];
#pragma unroll
    for (int mi = 0; mi < 4; ++mi) {
      const __bf16* p = aRow + (size_t)mi * 16 * N_IN + kb + lhi * 8;
      v8bf lo = *(const v8bf*)p;
      v8bf hi = *(const v8bf*)(p + 16);
      afrag[mi] = __builtin_shufflevector(lo, hi, 0, 1, 2, 3, 4, 5, 6, 7,
                                          8, 9, 10, 11, 12, 13, 14, 15);
    }
#pragma unroll
    for (int ni = 0; ni < 4; ++ni) {
      const __bf16* p = bRow + (size_t)ni * 16 * N_IN + kb + lhi * 8;
      v8bf lo = *(const v8bf*)p;
      v8bf hi = *(const v8bf*)(p + 16);
      bfrag[ni] = __builtin_shufflevector(lo, hi, 0, 1, 2, 3, 4, 5, 6, 7,
                                          8, 9, 10, 11, 12, 13, 14, 15);
    }
    if (kb + 32 < N_IN) {                       // pull next K slab toward WGP
      __builtin_prefetch(aRow + kb + 32, 0, 1); // -> global_prefetch_b8
      __builtin_prefetch(bRow + kb + 32, 0, 1);
    }
#pragma unroll
    for (int mi = 0; mi < 4; ++mi)
#pragma unroll
      for (int ni = 0; ni < 4; ++ni)
        c[mi][ni] = __builtin_amdgcn_wmma_f32_16x16x32_bf16(
            /*neg_a=*/false, afrag[mi], /*neg_b=*/false, bfrag[ni],
            /*c_mod=*/(short)0, c[mi][ni], /*reuse_a=*/false, /*reuse_b=*/false);
  }

  // D layout: lane holds column n = l15; VGPR r holds row m = lhi*8 + r.
#pragma unroll
  for (int mi = 0; mi < 4; ++mi) {
#pragma unroll
    for (int ni = 0; ni < 4; ++ni) {
      const int n  = nBase + ni * 16 + l15;
      const int m0 = mBase + mi * 16 + lhi * 8;
#pragma unroll
      for (int r = 0; r < 8; ++r)
        rf[(size_t)(m0 + r) * N_OUT + n] = c[mi][ni][r];
    }
  }
}

// ---------------------------------------------------------------- kernel 4 --
__device__ __forceinline__ float lif_step(float& v, float I) {
  v = 0.95f * v + I;                       // leaky integrate
  const float s = (v >= 1.0f) ? 1.0f : 0.0f;
  v = (s > 0.0f) ? 0.0f : v;               // hard reset
  return s;
}

// One thread per neuron; loads are independent of the v-chain so batching 4
// lets them pipeline under LOADcnt while the short VALU chain retires.
__global__ void snn_lif_scan(const float* __restrict__ rf, float* __restrict__ out) {
  const int o = blockIdx.x * blockDim.x + threadIdx.x;
  const float* p = rf + o;                 // rf[t][o], stride N_OUT
  float* q = out + (size_t)o * T_STEPS;    // out[o][t]
  float v = 0.0f;
  int t = 0;
  for (; t + 4 <= T_STEPS; t += 4) {
    const float i0 = p[(size_t)(t + 0) * N_OUT];
    const float i1 = p[(size_t)(t + 1) * N_OUT];
    const float i2 = p[(size_t)(t + 2) * N_OUT];
    const float i3 = p[(size_t)(t + 3) * N_OUT];
    const float s0 = lif_step(v, i0);
    const float s1 = lif_step(v, i1);
    const float s2 = lif_step(v, i2);
    const float s3 = lif_step(v, i3);
    q[t + 0] = s0; q[t + 1] = s1; q[t + 2] = s2; q[t + 3] = s3;
  }
  for (; t < T_STEPS; ++t)
    q[t] = lif_step(v, p[(size_t)t * N_OUT]);
}

// ---------------------------------------------------------------------------
extern "C" void kernel_launch(void* const* d_in, const int* in_sizes, int n_in,
                              void* d_out, int out_size, void* d_ws, size_t ws_size,
                              hipStream_t stream) {
  const float* in = (const float*)d_in[0];   // [64,64,4096] = [4096, 4096]
  const float* w  = (const float*)d_in[1];   // [4096, 4096]
  float* out = (float*)d_out;                // [4096, 4095]

  char* ws = (char*)d_ws;
  __bf16* Abf = (__bf16*)ws;                                  // 32 MB
  __bf16* Wbf = (__bf16*)(ws + ((size_t)32 << 20));           // 32 MB
  float*  rf  = (float*)(ws + ((size_t)64 << 20));            // 64 MB
  (void)in_sizes; (void)n_in; (void)out_size; (void)ws_size;

  // [1] transpose+convert stimulus (shift by 1 timestep, zero pad row)
  snn_build_A<<<dim3(T_TOT / 32, N_IN / 32), dim3(32, 8), 0, stream>>>(in, Abf);
  // [2] weights f32 -> bf16
  snn_convert_w<<<(N_OUT * (N_IN / 4)) / 256, 256, 0, stream>>>(w, Wbf);
  // [3] WMMA GEMM: rf[t][o]
  snn_gemm_bf16<<<dim3(N_OUT / 128, T_TOT / 256), 256, 0, stream>>>(Abf, Wbf, rf);
  // [4] sequential LIF scan
  snn_lif_scan<<<N_OUT / 256, 256, 0, stream>>>(rf, out);
}
